// StructuredAttentionLayer_89593017794932
// MI455X (gfx1250) — matrix operand order repaired and checked by reading, not compiled, not run on hardware
//
#include <hip/hip_runtime.h>
#include <hip/hip_bf16.h>

typedef __attribute__((ext_vector_type(16))) __bf16 v16bf;
typedef __attribute__((ext_vector_type(8)))  __bf16 v8bf;
typedef __attribute__((ext_vector_type(8)))  float  v8f;

#define BM 128
#define BN 128
#define BK 32
#define ASTR 40   // bf16 elements per LDS row (32 + 8 pad -> conflict-free ds_load_b128)

// H1 = relu(A[f32, MxK] @ W[f32, KxN] + bias), stored as bf16 for the next layer.
__global__ __launch_bounds__(256) void gemm1_relu_wmma(
    const float* __restrict__ A, const float* __restrict__ W,
    const float* __restrict__ bias, __bf16* __restrict__ C,
    int M, int K, int Nn) {
  __shared__ __align__(16) __bf16 As[BM * ASTR];
  __shared__ __align__(16) __bf16 Bs[BN * ASTR];

  const int t    = threadIdx.x;
  const int lane = t & 31;
  const int wid  = t >> 5;            // 8 waves
  const int wm   = wid >> 2;          // 0..1 -> 64 rows each
  const int wn   = wid & 3;           // 0..3 -> 32 cols each
  const int half = lane >> 4;
  const int l16  = lane & 15;

  const int blockM = blockIdx.x * BM;
  const int n0     = blockIdx.y * BN;

  v8f acc[4][2];
#pragma unroll
  for (int mt = 0; mt < 4; ++mt)
#pragma unroll
    for (int nt = 0; nt < 2; ++nt) acc[mt][nt] = (v8f){};

  for (int k0 = 0; k0 < K; k0 += BK) {
    // ---- stage A tile (128x32 f32 -> bf16): batch loads, then convert ----
    float4 va[4];
#pragma unroll
    for (int i = 0; i < 4; ++i) {
      int c   = t + i * 256;
      int row = c >> 3;
      int kc  = (c & 7) * 4;
      va[i] = *(const float4*)(A + (size_t)(blockM + row) * K + k0 + kc);
    }
#pragma unroll
    for (int i = 0; i < 4; ++i) {
      int c   = t + i * 256;
      int row = c >> 3;
      int kc  = (c & 7) * 4;
      As[row * ASTR + kc + 0] = (__bf16)va[i].x;
      As[row * ASTR + kc + 1] = (__bf16)va[i].y;
      As[row * ASTR + kc + 2] = (__bf16)va[i].z;
      As[row * ASTR + kc + 3] = (__bf16)va[i].w;
    }
    // one prefetch per 128B row-slab of the next K tile
    if (((t & 7) == 0) && (k0 + BK < K)) {
#pragma unroll
      for (int i = 0; i < 4; ++i) {
        int row = (t + i * 256) >> 3;
        __builtin_prefetch(A + (size_t)(blockM + row) * K + k0 + BK, 0, 1);
      }
    }
    // ---- stage B transposed: Bs[n][k] ----
#pragma unroll
    for (int i = 0; i < 16; ++i) {
      int c  = t + i * 256;
      int kk = c >> 7;
      int nn = c & 127;
      Bs[nn * ASTR + kk] = (__bf16)W[(size_t)(k0 + kk) * Nn + n0 + nn];
    }
    __syncthreads();

    v16bf bfrag[2];
#pragma unroll
    for (int nt = 0; nt < 2; ++nt) {
      int col = wn * 32 + nt * 16 + l16;
      const v8bf* bp = (const v8bf*)&Bs[col * ASTR + half * 16];
      v8bf lo = bp[0], hi = bp[1];
#pragma unroll
      for (int e = 0; e < 8; ++e) { bfrag[nt][e] = lo[e]; bfrag[nt][8 + e] = hi[e]; }
    }
#pragma unroll
    for (int mt = 0; mt < 4; ++mt) {
      int row = wm * 64 + mt * 16 + l16;
      v8bf lo = *(const v8bf*)&As[row * ASTR + half * 16];
      v8bf hi = *(const v8bf*)&As[row * ASTR + half * 16 + 8];
      v16bf afrag;
#pragma unroll
      for (int e = 0; e < 8; ++e) { afrag[e] = lo[e]; afrag[8 + e] = hi[e]; }
#pragma unroll
      for (int nt = 0; nt < 2; ++nt)
        acc[mt][nt] = __builtin_amdgcn_wmma_f32_16x16x32_bf16(
            false, afrag, false, bfrag[nt], (short)0, acc[mt][nt], false, false);
    }
    __syncthreads();
  }

#pragma unroll
  for (int mt = 0; mt < 4; ++mt)
#pragma unroll
    for (int nt = 0; nt < 2; ++nt) {
      int col  = n0 + wn * 32 + nt * 16 + l16;
      float bv = bias[col];
#pragma unroll
      for (int r = 0; r < 8; ++r) {
        int row = blockM + wm * 64 + mt * 16 + half * 8 + r;
        C[(size_t)row * Nn + col] = (__bf16)fmaxf(acc[mt][nt][r] + bv, 0.0f);
      }
    }
}

// Fused layers 2+3: un = relu(bf16 H1 @ W1 + b1) @ W2 + b2.
// Block tile 128x256 (full N), 8 waves: wave w owns rows [w*16, w*16+16) x all 256 cols
// -> each output row lives in one wave; layer-3 dot reduces with shfl, no H2 buffer.
// A tile is a pure bf16 copy -> async global->LDS (ASYNCcnt) path.
__global__ __launch_bounds__(256) void gemm2_un_wmma(
    const __bf16* __restrict__ A,   // [M,256] bf16
    const float* __restrict__ W1, const float* __restrict__ b1,
    const float* __restrict__ W2, const float* __restrict__ b2,
    float* __restrict__ un, int M) {
  constexpr int K = 256, NN = 256;
  __shared__ __align__(16) __bf16 As[BM * ASTR];
  __shared__ __align__(16) __bf16 Bs[NN * ASTR];
  __shared__ float sW2[NN * 2];
  __shared__ float sb1[NN];

  const int t    = threadIdx.x;
  const int lane = t & 31;
  const int wid  = t >> 5;            // 8 waves, 16 rows each
  const int half = lane >> 4;
  const int l16  = lane & 15;
  const int blockM = blockIdx.x * BM;

  for (int i = t; i < NN * 2; i += 256) sW2[i] = W2[i];
  sb1[t] = b1[t];
  const float bb0 = b2[0], bb1 = b2[1];

  v8f acc[16];
#pragma unroll
  for (int nt = 0; nt < 16; ++nt) acc[nt] = (v8f){};

  for (int k0 = 0; k0 < K; k0 += BK) {
    // ---- stage A tile (128x32 bf16) via async 16B global->LDS copies ----
#pragma unroll
    for (int i = 0; i < 2; ++i) {
      int c   = t + i * 256;          // 0..511 16B chunks
      int row = c >> 2;
      int kc  = (c & 3) * 8;
      unsigned int  lds_off = (unsigned int)(uintptr_t)&As[row * ASTR + kc];
      unsigned long long gaddr =
          (unsigned long long)(uintptr_t)(A + (size_t)(blockM + row) * K + k0 + kc);
      asm volatile("global_load_async_to_lds_b128 %0, %1, off"
                   :: "v"(lds_off), "v"(gaddr) : "memory");
    }
    // ---- stage B transposed: W1 tile 32x256 f32 -> Bs[n][k] bf16 ----
#pragma unroll
    for (int i = 0; i < 32; ++i) {
      int c  = t + i * 256;           // 0..8191
      int kk = c >> 8;
      int nn = c & 255;
      Bs[nn * ASTR + kk] = (__bf16)W1[(size_t)(k0 + kk) * NN + nn];
    }
    asm volatile("s_wait_asynccnt 0" ::: "memory");
    __syncthreads();

    int row = wid * 16 + l16;
    v8bf alo = *(const v8bf*)&As[row * ASTR + half * 16];
    v8bf ahi = *(const v8bf*)&As[row * ASTR + half * 16 + 8];
    v16bf afrag;
#pragma unroll
    for (int e = 0; e < 8; ++e) { afrag[e] = alo[e]; afrag[8 + e] = ahi[e]; }

#pragma unroll
    for (int nt = 0; nt < 16; ++nt) {
      int col = nt * 16 + l16;
      const v8bf* bp = (const v8bf*)&Bs[col * ASTR + half * 16];
      v8bf lo = bp[0], hi = bp[1];
      v16bf bfrag;
#pragma unroll
      for (int e = 0; e < 8; ++e) { bfrag[e] = lo[e]; bfrag[8 + e] = hi[e]; }
      acc[nt] = __builtin_amdgcn_wmma_f32_16x16x32_bf16(
          false, afrag, false, bfrag, (short)0, acc[nt], false, false);
    }
    __syncthreads();
  }

  // epilogue: relu(acc + b1) then dot with W2 cols, reduce over the 16 lanes
  // that hold one row (xor offsets < 16 stay inside each half).
#pragma unroll
  for (int r = 0; r < 8; ++r) {
    float p0 = 0.0f, p1 = 0.0f;
#pragma unroll
    for (int nt = 0; nt < 16; ++nt) {
      int col = nt * 16 + l16;
      float v = fmaxf(acc[nt][r] + sb1[col], 0.0f);
      p0 += v * sW2[col * 2 + 0];
      p1 += v * sW2[col * 2 + 1];
    }
#pragma unroll
    for (int off = 8; off >= 1; off >>= 1) {
      p0 += __shfl_xor(p0, off, 32);
      p1 += __shfl_xor(p1, off, 32);
    }
    if (l16 == 0) {
      int rowg = blockM + wid * 16 + half * 8 + r;
      un[(size_t)rowg * 2 + 0] = p0 + bb0;
      un[(size_t)rowg * 2 + 1] = p1 + bb1;
    }
  }
}

// relation encoder: ctx[10752,5] -> relu(@R0[5,64]+r0) @ R1[64,3]+r1 -> bp4[10752,2,2]
__global__ __launch_bounds__(256) void rel_kernel(
    const float* __restrict__ ctx, const float* __restrict__ R0,
    const float* __restrict__ r0, const float* __restrict__ R1,
    const float* __restrict__ r1, float* __restrict__ bp4, int ROWS) {
  __shared__ float sR0[5 * 64];
  __shared__ float sr0[64];
  __shared__ float sR1[64 * 3];
  __shared__ float sr1[3];
  int t = threadIdx.x;
  for (int i = t; i < 320; i += 256) sR0[i] = R0[i];
  for (int i = t; i < 192; i += 256) sR1[i] = R1[i];
  if (t < 64) sr0[t] = r0[t];
  if (t < 3)  sr1[t] = r1[t];
  __syncthreads();
  int row = blockIdx.x * 256 + t;
  if (row >= ROWS) return;
  float c[5];
#pragma unroll
  for (int x = 0; x < 5; ++x) c[x] = ctx[(size_t)row * 5 + x];
  float y0 = sr1[0], y1 = sr1[1], y2 = sr1[2];
#pragma unroll 4
  for (int j = 0; j < 64; ++j) {
    float h = sr0[j];
#pragma unroll
    for (int x = 0; x < 5; ++x) h += c[x] * sR0[x * 64 + j];
    h = fmaxf(h, 0.0f);
    y0 += h * sR1[j * 3 + 0];
    y1 += h * sR1[j * 3 + 1];
    y2 += h * sR1[j * 3 + 2];
  }
  float* o = bp4 + (size_t)row * 4;   // Mmap folded: [[y0,y1],[y1,y2]]
  o[0] = y0; o[1] = y1; o[2] = y1; o[3] = y2;
}

// joint factor graph: block = batch b (512 blocks), 128 threads = 128 states
__global__ __launch_bounds__(128) void joint_kernel(
    const float* __restrict__ un, const float* __restrict__ bp4,
    float* __restrict__ marg, float* __restrict__ jlp) {
  constexpr int NE = 7, NPAIR = 21, NN = 32, BSZ = 512;
  constexpr int PI[NPAIR] = {0,0,0,0,0,0,1,1,1,1,1,2,2,2,2,3,3,3,4,4,5};
  constexpr int PJ[NPAIR] = {1,2,3,4,5,6,2,3,4,5,6,3,4,5,6,4,5,6,5,6,6};

  __shared__ float sh_bp[NPAIR * 4];
  __shared__ float sh_un[NE * 2];
  __shared__ float sh_max[4];
  __shared__ float sh_red[4 * 8];
  __shared__ float sh_tot[8];

  const int b = blockIdx.x;
  const int s = threadIdx.x;
  const int lane = s & 31, wv = s >> 5;

  if (s < NPAIR * 4) sh_bp[s] = bp4[(size_t)b * NPAIR * 4 + s];
  __syncthreads();

  int bit[NE];
#pragma unroll
  for (int i = 0; i < NE; ++i) bit[i] = (s >> (6 - i)) & 1;

  float pairSum = 0.0f;
#pragma unroll
  for (int p = 0; p < NPAIR; ++p)
    pairSum += sh_bp[p * 4 + bit[PI[p]] * 2 + bit[PJ[p]]];

  for (int n = 0; n < NN; ++n) {
    const size_t nb = (size_t)n * BSZ + b;
    if (s < NE * 2) sh_un[s] = un[nb * 14 + s];
    __syncthreads();

    float val = pairSum;
#pragma unroll
    for (int i = 0; i < NE; ++i) val += sh_un[i * 2 + bit[i]];

    float m = val;
#pragma unroll
    for (int off = 16; off >= 1; off >>= 1) m = fmaxf(m, __shfl_xor(m, off, 32));
    if (lane == 0) sh_max[wv] = m;
    __syncthreads();
    m = fmaxf(fmaxf(sh_max[0], sh_max[1]), fmaxf(sh_max[2], sh_max[3]));

    float e = __expf(val - m);
    float f[8];
    f[0] = e;
#pragma unroll
    for (int i = 0; i < NE; ++i) f[1 + i] = bit[i] ? e : 0.0f;
#pragma unroll
    for (int off = 16; off >= 1; off >>= 1)
#pragma unroll
      for (int k = 0; k < 8; ++k) f[k] += __shfl_xor(f[k], off, 32);
    if (lane == 0)
#pragma unroll
      for (int k = 0; k < 8; ++k) sh_red[wv * 8 + k] = f[k];
    __syncthreads();
    if (s < 8) sh_tot[s] = sh_red[s] + sh_red[8 + s] + sh_red[16 + s] + sh_red[24 + s];
    __syncthreads();

    const float S = sh_tot[0];
    jlp[nb * 128 + s] = val - (m + __logf(S));
    if (s < NE) {
      float S1 = sh_tot[1 + s];
      marg[nb * NE + s] = __logf(S1) - __logf(S - S1);
    }
    __syncthreads();
  }
}

extern "C" void kernel_launch(void* const* d_in, const int* in_sizes, int n_in,
                              void* d_out, int out_size, void* d_ws, size_t ws_size,
                              hipStream_t stream) {
  const float* input = (const float*)d_in[0];
  const float* ctx   = (const float*)d_in[1];
  // d_in[2] lang_input, d_in[3] num_markables: unused by reference math
  const float* W0 = (const float*)d_in[4];
  const float* b0 = (const float*)d_in[5];
  const float* W1 = (const float*)d_in[6];
  const float* b1 = (const float*)d_in[7];
  const float* W2 = (const float*)d_in[8];
  const float* b2 = (const float*)d_in[9];
  const float* R0 = (const float*)d_in[10];
  const float* r0 = (const float*)d_in[11];
  const float* R1 = (const float*)d_in[12];
  const float* r1 = (const float*)d_in[13];

  const int M = 32 * 512 * 7;     // 114688 rows
  const int RELROWS = 512 * 21;   // 10752

  float* out  = (float*)d_out;
  float* marg = out;                               // [32,512,7]
  float* jlp  = out + (size_t)32 * 512 * 7;        // [32,512,128]

  __bf16* H1 = (__bf16*)d_ws;                      // M*256 bf16
  float* un  = (float*)((char*)d_ws + (size_t)M * 256 * sizeof(__bf16));  // M*2 f32
  float* bp4 = un + (size_t)M * 2;                 // 10752*4 f32

  dim3 g1(M / BM, 256 / BN);
  gemm1_relu_wmma<<<g1, 256, 0, stream>>>(input, W0, b0, H1, M, 512, 256);
  gemm2_un_wmma<<<M / BM, 256, 0, stream>>>(H1, W1, b1, W2, b2, un, M);
  rel_kernel<<<(RELROWS + 255) / 256, 256, 0, stream>>>(ctx, R0, r0, R1, r1, bp4, RELROWS);
  joint_kernel<<<512, 128, 0, stream>>>(un, bp4, marg, jlp);
}
